// WordRRN_51539608191
// MI455X (gfx1250) — matrix operand order repaired and checked by reading, not compiled, not run on hardware
//
#include <hip/hip_runtime.h>
#include <hip/hip_bf16.h>

typedef __attribute__((ext_vector_type(16))) _Float16 v16h;
typedef __attribute__((ext_vector_type(8)))  float    v8f;

#define SEQ    2048
#define BATCH  64
#define HID    256
#define EMB    128
#define VOCAB  50257
#define NTOK   (BATCH * SEQ)   // 131072 tokens

// ---------------------------------------------------------------------------
// WMMA fragment helpers (wave32, 16x16x32 f16 -> f32)
//
// A (16x32, f16): lane&15 = row M; lanes 0-15 carry K = k0+0..7 (elems 0..7)
//   and K = k0+16..23 (elems 8..15); lanes 16-31 carry K = k0+8..15 / k0+24..31.
// B (32x16, f16): lane&15 = col N; lanes 0-15 carry K = k0+0..7 (elems 0..7)
//   and K = k0+16..23 (elems 8..15); lanes 16-31 carry K = k0+8..15 / k0+24..31.
// C/D (16x16, f32): elem v -> row M = v + (lane>=16 ? 8 : 0), col N = lane&15.
// ---------------------------------------------------------------------------

__device__ __forceinline__ v16h load_a_f16(const _Float16* __restrict__ A,
                                           int row0, int k0) {
  const int lane = threadIdx.x & 31;
  const _Float16* p = A + (size_t)(row0 + (lane & 15)) * HID + k0 + ((lane & 16) ? 8 : 0);
  v16h a;
  ((float4*)&a)[0] = *(const float4*)(p);       // elems 0..7
  ((float4*)&a)[1] = *(const float4*)(p + 16);  // elems 8..15
  return a;
}

// Map (k,n) of a row-major KxHID weight to the B-fragment-packed offset
// (fragments stored contiguously: frag = kt*16+nt, 32 lanes x 16 halves).
__device__ __forceinline__ size_t b_swz(int k, int n) {
  const int kt = k >> 5, kk = k & 31, nt = n >> 4, nn = n & 15;
  const int lane = nn + ((kk & 8) << 1);
  const int elem = (kk & 7) | ((kk & 16) >> 1);
  return ((size_t)(kt * 16 + nt) * 32 + lane) * 16 + elem;
}

// ---------------------------------------------------------------------------
// Async global -> LDS staging of the fragment-packed weight tile.
// Uses GLOBAL_LOAD_ASYNC_TO_LDS_B128 (ASYNCcnt) to avoid the VGPR round-trip;
// generic shared pointers truncate to the LDS byte offset (flat aperture).
// ---------------------------------------------------------------------------
__device__ __forceinline__ void stage_b_async(const _Float16* __restrict__ Bsw,
                                              char* smem, int bytes) {
  const char* g = (const char*)Bsw;
  const unsigned lbase = (unsigned)(uintptr_t)smem;
  for (int i = threadIdx.x * 16; i < bytes; i += 256 * 16) {
    asm volatile("global_load_async_to_lds_b128 %0, %1, off"
                 :: "v"(lbase + (unsigned)i), "v"(g + i) : "memory");
  }
  asm volatile("s_wait_asynccnt 0x0" ::: "memory");
}

// ---------------------------------------------------------------------------
// Weight convert + swizzle: f32 row-major [K x 256] -> f16 fragment order.
// ---------------------------------------------------------------------------
__global__ __launch_bounds__(256) void cvt_swizzle(const float* __restrict__ src,
                                                   _Float16* __restrict__ dst,
                                                   int total) {
  const int idx = blockIdx.x * 256 + threadIdx.x;
  if (idx >= total) return;
  const int k = idx >> 8, n = idx & 255;
  dst[b_swz(k, n)] = (_Float16)src[idx];
}

// ---------------------------------------------------------------------------
// Embedding gather + input FC:  base = relu(emb[word] @ W_in + b_in)
// K = 128 (4 k-steps).  Writes base and h (== base) as f16.
// ---------------------------------------------------------------------------
__global__ __launch_bounds__(256) void embed_fc(const int* __restrict__ words,
                                                const float* __restrict__ emb,
                                                const _Float16* __restrict__ Bsw,
                                                const float* __restrict__ bias,
                                                _Float16* __restrict__ baseH,
                                                _Float16* __restrict__ hH) {
  extern __shared__ char smem[];
  stage_b_async(Bsw, smem, EMB * HID * 2);
  __syncthreads();

  const int lane = threadIdx.x & 31;
  const int wave = threadIdx.x >> 5;
  const int row0 = blockIdx.x * 128 + wave * 16;
  const int r = row0 + (lane & 15);
  const float* erow = emb + (size_t)words[r] * EMB;
  const int kofs = (lane & 16) ? 8 : 0;

  const v8f vz = {0.f, 0.f, 0.f, 0.f, 0.f, 0.f, 0.f, 0.f};
  v8f acc[16];
#pragma unroll
  for (int nt = 0; nt < 16; ++nt) acc[nt] = vz;

#pragma unroll
  for (int kt = 0; kt < 4; ++kt) {
    const float* p = erow + kt * 32 + kofs;
    v16h a;
#pragma unroll
    for (int t = 0; t < 8; ++t) a[t] = (_Float16)p[t];
#pragma unroll
    for (int t = 0; t < 8; ++t) a[8 + t] = (_Float16)p[16 + t];
#pragma unroll
    for (int nt = 0; nt < 16; ++nt) {
      v16h b = *(const v16h*)(smem + (size_t)(kt * 16 + nt) * 1024 + lane * 32);
      acc[nt] = __builtin_amdgcn_wmma_f32_16x16x32_f16(false, a, false, b,
                                                       (short)0, acc[nt], false, false);
    }
  }

  const int hi8 = (lane & 16) ? 8 : 0;
  const int nn = lane & 15;
#pragma unroll
  for (int nt = 0; nt < 16; ++nt) {
    const int col = nt * 16 + nn;
    const float bv = bias[col];
#pragma unroll
    for (int v = 0; v < 8; ++v) {
      const int row = row0 + v + hi8;
      const size_t idx = (size_t)row * HID + col;
      const float val = fmaxf(acc[nt][v] + bv, 0.f);
      const _Float16 hv = (_Float16)val;
      baseH[idx] = hv;
      hH[idx] = hv;
    }
  }
}

// ---------------------------------------------------------------------------
// Generic 131072 x 256 x 256 WMMA GEMM with fused epilogue.
//   OUTMODE: 0 = f32 store, 1 = f32 accumulate, 2 = f16 store
//   MASK:    0 = none, 1 = zero rows with s==0, 2 = zero rows with s==SEQ-1
// ---------------------------------------------------------------------------
template <bool RELU, bool BIAS, int NADD, int OUTMODE, int MASK>
__global__ __launch_bounds__(256) void wmma_gemm256(const _Float16* __restrict__ A,
                                                    const _Float16* __restrict__ Bsw,
                                                    const float* __restrict__ bias,
                                                    const float* __restrict__ add0,
                                                    const float* __restrict__ add1,
                                                    float* __restrict__ outF,
                                                    _Float16* __restrict__ outH) {
  extern __shared__ char smem[];
  stage_b_async(Bsw, smem, HID * HID * 2);
  __syncthreads();

  const int lane = threadIdx.x & 31;
  const int wave = threadIdx.x >> 5;
  const int row0 = blockIdx.x * 128 + wave * 16;

  const v8f vz = {0.f, 0.f, 0.f, 0.f, 0.f, 0.f, 0.f, 0.f};
  v8f acc[16];
#pragma unroll
  for (int nt = 0; nt < 16; ++nt) acc[nt] = vz;

#pragma unroll
  for (int kt = 0; kt < 8; ++kt) {
    v16h a = load_a_f16(A, row0, kt * 32);
#pragma unroll
    for (int nt = 0; nt < 16; ++nt) {
      v16h b = *(const v16h*)(smem + (size_t)(kt * 16 + nt) * 1024 + lane * 32);
      acc[nt] = __builtin_amdgcn_wmma_f32_16x16x32_f16(false, a, false, b,
                                                       (short)0, acc[nt], false, false);
    }
  }

  const int hi8 = (lane & 16) ? 8 : 0;
  const int nn = lane & 15;
#pragma unroll
  for (int nt = 0; nt < 16; ++nt) {
    const int col = nt * 16 + nn;
    const float bv = BIAS ? bias[col] : 0.f;
#pragma unroll
    for (int v = 0; v < 8; ++v) {
      const int row = row0 + v + hi8;
      const size_t idx = (size_t)row * HID + col;
      float val = acc[nt][v] + bv;
      if (NADD >= 1) val += add0[idx];
      if (NADD >= 2) val += add1[idx];
      if (RELU) val = fmaxf(val, 0.f);
      if (MASK == 1 && (row & (SEQ - 1)) == 0) val = 0.f;
      if (MASK == 2 && (row & (SEQ - 1)) == (SEQ - 1)) val = 0.f;
      if (OUTMODE == 0) outF[idx] = val;
      else if (OUTMODE == 1) outF[idx] += val;
      else outH[idx] = (_Float16)val;
    }
  }
}

// ---------------------------------------------------------------------------
// tL[s] = relu(U[s] + V[s-1] + b_m1)  (0 at s==0)
// tR[s] = relu(U[s] + V[s+1] + b_m1)  (0 at s==SEQ-1)
// ---------------------------------------------------------------------------
__global__ __launch_bounds__(256) void make_t_kernel(const _Float16* __restrict__ U,
                                                     const _Float16* __restrict__ V,
                                                     const float* __restrict__ b_m1,
                                                     _Float16* __restrict__ tL,
                                                     _Float16* __restrict__ tR) {
  const size_t i = (size_t)blockIdx.x * 256 + threadIdx.x;
  const int col = (int)(i & (HID - 1));
  const int tok = (int)(i >> 8);
  const int s = tok & (SEQ - 1);
  const float u = (float)U[i] + b_m1[col];
  float l = 0.f, r = 0.f;
  if (s > 0)       l = fmaxf(u + (float)V[i - HID], 0.f);
  if (s < SEQ - 1) r = fmaxf(u + (float)V[i + HID], 0.f);
  tL[i] = (_Float16)l;
  tR[i] = (_Float16)r;
}

// ---------------------------------------------------------------------------
// Mean pool over sequence:  hpool[b][c] = mean_s h[b,s,c]
// ---------------------------------------------------------------------------
__global__ __launch_bounds__(256) void pool_kernel(const _Float16* __restrict__ hH,
                                                   float* __restrict__ hpool) {
  const int b = blockIdx.x, c = threadIdx.x;
  const _Float16* p = hH + (size_t)b * SEQ * HID + c;
  float s = 0.f;
  for (int t = 0; t < SEQ; ++t) s += (float)p[(size_t)t * HID];
  hpool[b * HID + c] = s * (1.0f / SEQ);
}

// ---------------------------------------------------------------------------
// Classifier:  out[b][v] = hpool[b] . W_out[:,v] + b_out[v]
// ---------------------------------------------------------------------------
__global__ __launch_bounds__(256) void logits_kernel(const float* __restrict__ hpool,
                                                     const float* __restrict__ W_out,
                                                     const float* __restrict__ b_out,
                                                     float* __restrict__ out) {
  __shared__ float hrow[HID];
  const int b = blockIdx.y;
  hrow[threadIdx.x] = hpool[b * HID + threadIdx.x];
  __syncthreads();
  const int v = blockIdx.x * 256 + threadIdx.x;
  if (v < VOCAB) {
    float a = b_out[v];
#pragma unroll 4
    for (int k = 0; k < HID; ++k) a = fmaf(hrow[k], W_out[(size_t)k * VOCAB + v], a);
    out[(size_t)b * VOCAB + v] = a;
  }
}

// ---------------------------------------------------------------------------
// In-place log-softmax per row (one block per batch row).
// ---------------------------------------------------------------------------
__global__ __launch_bounds__(256) void logsoftmax_kernel(float* __restrict__ out) {
  __shared__ float red[256];
  float* row = out + (size_t)blockIdx.x * VOCAB;

  float mx = -3.402823e38f;
  for (int v = threadIdx.x; v < VOCAB; v += 256) mx = fmaxf(mx, row[v]);
  red[threadIdx.x] = mx;
  __syncthreads();
  for (int s = 128; s > 0; s >>= 1) {
    if (threadIdx.x < s) red[threadIdx.x] = fmaxf(red[threadIdx.x], red[threadIdx.x + s]);
    __syncthreads();
  }
  mx = red[0];
  __syncthreads();

  float sm = 0.f;
  for (int v = threadIdx.x; v < VOCAB; v += 256) sm += expf(row[v] - mx);
  red[threadIdx.x] = sm;
  __syncthreads();
  for (int s = 128; s > 0; s >>= 1) {
    if (threadIdx.x < s) red[threadIdx.x] += red[threadIdx.x + s];
    __syncthreads();
  }
  const float lse = mx + logf(red[0]);
  __syncthreads();

  for (int v = threadIdx.x; v < VOCAB; v += 256) row[v] -= lse;
}

// ---------------------------------------------------------------------------
// Orchestration
// ---------------------------------------------------------------------------
extern "C" void kernel_launch(void* const* d_in, const int* in_sizes, int n_in,
                              void* d_out, int out_size, void* d_ws, size_t ws_size,
                              hipStream_t stream) {
  const int*   words = (const int*)d_in[0];
  const float* emb   = (const float*)d_in[1];
  const float* W_in  = (const float*)d_in[2];
  const float* b_in  = (const float*)d_in[3];
  const float* W_m1  = (const float*)d_in[4];
  const float* b_m1  = (const float*)d_in[5];
  const float* W_m2  = (const float*)d_in[6];
  const float* b_m2  = (const float*)d_in[7];
  const float* W_up  = (const float*)d_in[8];
  const float* b_up  = (const float*)d_in[9];
  const float* W_out = (const float*)d_in[10];
  const float* b_out = (const float*)d_in[11];
  float* out = (float*)d_out;

  char* ws = (char*)d_ws;
  size_t o = 0;
  auto take = [&](size_t bytes) -> char* {
    char* p = ws + o;
    o += (bytes + 255) & ~(size_t)255;
    return p;
  };

  const size_t actH = (size_t)NTOK * HID * 2;  // f16 activation plane
  const size_t actF = (size_t)NTOK * HID * 4;  // f32 activation plane

  _Float16* w_in16 = (_Float16*)take((size_t)EMB * HID * 2);
  _Float16* wm1top = (_Float16*)take((size_t)HID * HID * 2);
  _Float16* wm1bot = (_Float16*)take((size_t)HID * HID * 2);
  _Float16* wm2    = (_Float16*)take((size_t)HID * HID * 2);
  _Float16* wuptop = (_Float16*)take((size_t)HID * HID * 2);
  _Float16* wupbot = (_Float16*)take((size_t)HID * HID * 2);
  _Float16* baseH  = (_Float16*)take(actH);
  _Float16* hH     = (_Float16*)take(actH);
  float*    Cb     = (float*)take(actF);
  char*     uvm    = take(actF);               // U|V (f16) aliased with m (f32)
  _Float16* U16    = (_Float16*)uvm;
  _Float16* V16    = (_Float16*)(uvm + actH);
  float*    m      = (float*)uvm;
  _Float16* tL     = (_Float16*)take(actH);
  _Float16* tR     = (_Float16*)take(actH);
  float*    hpool  = (float*)take((size_t)BATCH * HID * 4);

  const int gemmGrid = NTOK / 128;                  // 1024 blocks, 8 waves each
  const size_t ldsB  = (size_t)HID * HID * 2;       // 128 KB weight tile

  // --- one-time: weight convert + swizzle ---
  cvt_swizzle<<<EMB, 256, 0, stream>>>(W_in, w_in16, EMB * HID);
  cvt_swizzle<<<HID, 256, 0, stream>>>(W_m1, wm1top, HID * HID);
  cvt_swizzle<<<HID, 256, 0, stream>>>(W_m1 + HID * HID, wm1bot, HID * HID);
  cvt_swizzle<<<HID, 256, 0, stream>>>(W_m2, wm2, HID * HID);
  cvt_swizzle<<<HID, 256, 0, stream>>>(W_up, wuptop, HID * HID);
  cvt_swizzle<<<HID, 256, 0, stream>>>(W_up + HID * HID, wupbot, HID * HID);

  // --- base = relu(emb @ W_in + b_in); h = base ---
  embed_fc<<<gemmGrid, 256, (size_t)EMB * HID * 2, stream>>>(words, emb, w_in16,
                                                             b_in, baseH, hH);

  // --- Cb = base @ W_up_bot + b_up (step-invariant) ---
  wmma_gemm256<false, true, 0, 0, 0><<<gemmGrid, 256, ldsB, stream>>>(
      baseH, wupbot, b_up, nullptr, nullptr, Cb, nullptr);

  // --- 3 message-passing steps ---
  for (int step = 0; step < 3; ++step) {
    // U = h @ Wm1_top ; V = h @ Wm1_bot   (f16)
    wmma_gemm256<false, false, 0, 2, 0><<<gemmGrid, 256, ldsB, stream>>>(
        hH, wm1top, nullptr, nullptr, nullptr, nullptr, U16);
    wmma_gemm256<false, false, 0, 2, 0><<<gemmGrid, 256, ldsB, stream>>>(
        hH, wm1bot, nullptr, nullptr, nullptr, nullptr, V16);

    // tL / tR from shifted sums
    make_t_kernel<<<(NTOK * HID) / 256, 256, 0, stream>>>(U16, V16, b_m1, tL, tR);

    // m  = mask1(relu(tL @ Wm2 + b_m2))          (f32 store; overwrites U|V)
    wmma_gemm256<true, true, 0, 0, 1><<<gemmGrid, 256, ldsB, stream>>>(
        tL, wm2, b_m2, nullptr, nullptr, m, nullptr);
    // m += mask2(relu(tR @ Wm2 + b_m2))
    wmma_gemm256<true, true, 0, 1, 2><<<gemmGrid, 256, ldsB, stream>>>(
        tR, wm2, b_m2, nullptr, nullptr, m, nullptr);

    // h = relu(h @ W_up_top + Cb + m)            (in-place f16)
    wmma_gemm256<true, false, 2, 2, 0><<<gemmGrid, 256, ldsB, stream>>>(
        hH, wuptop, nullptr, Cb, m, nullptr, hH);
  }

  // --- pool, classifier, log-softmax ---
  pool_kernel<<<BATCH, 256, 0, stream>>>(hH, hpool);
  logits_kernel<<<dim3((VOCAB + 255) / 256, BATCH), 256, 0, stream>>>(hpool, W_out,
                                                                      b_out, out);
  logsoftmax_kernel<<<BATCH, 256, 0, stream>>>(out);
}
